// ModifiedCogVideoXBlock_87840671138311
// MI455X (gfx1250) — matrix operand order repaired and checked by reading, not compiled, not run on hardware
//
#include <hip/hip_runtime.h>
#include <stdint.h>

#define HEADS     16
#define HEAD_DIM  64
#define DIM       1024
#define TEXT_LEN  226
#define NCPT      8
#define FRAMES    4
#define PATCHES   384
#define IMG_LEN   1536
#define S_TOT     1762
#define S_CAT     1544
#define TIME_DIM  512
#define FF_DIM    4096
#define NB        2
#define M_ROWS    (NB*S_TOT)   /* 3524 */
#define M_PAD     3584         /* ceil to 64 for fat GEMM tiles */
#define CM_PAD    64           /* padded concept-row count */
#define S_PAD     1792         /* padded key length (mult of 32) */
#define QTILES    111          /* ceil(1762/16) */

typedef _Float16 f16_t;
typedef __attribute__((ext_vector_type(16))) _Float16 v16h;
typedef __attribute__((ext_vector_type(8)))  _Float16 v8h;
typedef __attribute__((ext_vector_type(8)))  float    v8f;

#define SHUF16(lo,hi) __builtin_shufflevector(lo, hi, 0,1,2,3,4,5,6,7,8,9,10,11,12,13,14,15)

__device__ __forceinline__ v8f wmma16(v16h a, v16h b, v8f c){
    return __builtin_amdgcn_wmma_f32_16x16x32_f16(false, a, false, b, (short)0, c, false, false);
}

// ---------------------------------------------------------------- utilities
__global__ void k_f32_to_f16(const float* __restrict__ x, f16_t* __restrict__ y, size_t n){
    size_t i = (size_t)blockIdx.x*256 + threadIdx.x;
    if(i < n) y[i] = (f16_t)x[i];
}
__global__ void k_fill_h(f16_t* p, size_t n){
    size_t i = (size_t)blockIdx.x*256 + threadIdx.x;
    if(i < n) p[i] = (f16_t)0.f;
}

// ------------------------------------------- s = silu(temb) @ W + b  (B,6144)
__global__ void k_modproj(const float* __restrict__ temb, const float* __restrict__ W,
                          const float* __restrict__ bias, float* __restrict__ out){
    __shared__ float sm[TIME_DIM];
    int b = blockIdx.y;
    for(int i = threadIdx.x; i < TIME_DIM; i += 256){
        float t = temb[b*TIME_DIM + i];
        sm[i] = t / (1.f + __expf(-t));
    }
    __syncthreads();
    int n = blockIdx.x*256 + threadIdx.x;           // < 6144
    float acc = bias[n];
    for(int k = 0; k < TIME_DIM; ++k) acc += sm[k]*W[(size_t)k*(6*DIM) + n];
    out[(size_t)b*6*DIM + n] = acc;
}

// --------------------------------- layernorm + adaLN modulation -> f16 rows
__global__ void k_ln_mod(const float* __restrict__ x, const float* __restrict__ sbuf,
                         const float* __restrict__ lnw, const float* __restrict__ lnb,
                         f16_t* __restrict__ out, int S, int shift_c, int scale_c,
                         int out_off, int out_bstride){
    __shared__ float r1[256], r2[256];
    int row = blockIdx.x, b = blockIdx.y;
    const float* xp = x + ((size_t)(b*S + row))*DIM;
    float v[4]; float s = 0.f, s2 = 0.f;
    #pragma unroll
    for(int j = 0; j < 4; ++j){
        float t = xp[threadIdx.x + 256*j]; v[j] = t; s += t; s2 += t*t;
    }
    r1[threadIdx.x] = s; r2[threadIdx.x] = s2; __syncthreads();
    for(int st = 128; st > 0; st >>= 1){
        if(threadIdx.x < st){ r1[threadIdx.x] += r1[threadIdx.x+st]; r2[threadIdx.x] += r2[threadIdx.x+st]; }
        __syncthreads();
    }
    float mu  = r1[0]*(1.f/DIM);
    float var = r2[0]*(1.f/DIM) - mu*mu;
    float inv = rsqrtf(var + 1e-5f);
    f16_t* op = out + ((size_t)(b*out_bstride + out_off + row))*DIM;
    const float* sh = sbuf + (size_t)b*6*DIM + shift_c*DIM;
    const float* sc = sbuf + (size_t)b*6*DIM + scale_c*DIM;
    #pragma unroll
    for(int j = 0; j < 4; ++j){
        int d = threadIdx.x + 256*j;
        float ln = (v[j]-mu)*inv*lnw[d] + lnb[d];
        op[d] = (f16_t)(ln*(1.f + sc[d]) + sh[d]);
    }
}

// ------------------------------------- WMMA GEMM, 64x64 block tile, 4 waves
// Each wave: one 16-col strip, 4 M-subtiles -> 4 wmma per K-step.
// EPI 0: f32 row-major + bias | EPI 1: scatter (b,h,s,d) f32 | EPI 2: gelu->f16
template<int EPI>
__global__ __launch_bounds__(128)
void k_gemm(const f16_t* __restrict__ A, const f16_t* __restrict__ Bw,
            const float* __restrict__ bias, void* __restrict__ outp,
            int M_valid, int N, int K, int seqlen){
    __shared__ alignas(32) f16_t lsA[64*32];
    __shared__ alignas(32) f16_t lsB[32*64];
    const int tid  = threadIdx.x;
    const int lane = tid & 31, wv = tid >> 5;
    const int g = lane >> 4, l15 = lane & 15;
    const int rowbase = blockIdx.y*64, colbase = blockIdx.x*64;
    v8f acc[4];
    #pragma unroll
    for(int ms = 0; ms < 4; ++ms)
        #pragma unroll
        for(int i = 0; i < 8; ++i) acc[ms][i] = 0.f;
    const int nk = K >> 5;
    for(int kk = 0; kk < nk; ++kk){
        {   // stage A tile 64x32 (2048 halfs)
            int r = tid >> 1, c = (tid & 1)*16;
            const f16_t* src = A + (size_t)(rowbase + r)*K + kk*32 + c;
            *(v8h*)(lsA + r*32 + c)     = *(const v8h*)src;
            *(v8h*)(lsA + r*32 + c + 8) = *(const v8h*)(src + 8);
        }
        {   // stage B tile 32x64 (2048 halfs)
            int r = tid >> 2, c = (tid & 3)*16;
            const f16_t* src = Bw + (size_t)(kk*32 + r)*N + colbase + c;
            *(v8h*)(lsB + r*64 + c)     = *(const v8h*)src;
            *(v8h*)(lsB + r*64 + c + 8) = *(const v8h*)(src + 8);
        }
        if(kk + 1 < nk){
            __builtin_prefetch(Bw + (size_t)(kk*32 + 32)*N + colbase + (tid & 3)*16, 0, 1);
            __builtin_prefetch(A + (size_t)(rowbase + (tid >> 1))*K + kk*32 + 32, 0, 1);
        }
        __syncthreads();
        v16h bfr;
        #pragma unroll
        for(int j = 0; j < 16; ++j) bfr[j] = lsB[((j & 7) + 8*g + 16*(j >> 3))*64 + wv*16 + l15];
        #pragma unroll
        for(int ms = 0; ms < 4; ++ms){
            v8h alo = *(const v8h*)(lsA + (ms*16 + l15)*32 + 8*g);
            v8h ahi = *(const v8h*)(lsA + (ms*16 + l15)*32 + 16 + 8*g);
            v16h a = SHUF16(alo, ahi);
            acc[ms] = wmma16(a, bfr, acc[ms]);
        }
        __syncthreads();
    }
    const int n = colbase + wv*16 + l15;
    if constexpr (EPI == 0){
        float* O = (float*)outp; float bb = bias[n];
        #pragma unroll
        for(int ms = 0; ms < 4; ++ms)
            #pragma unroll
            for(int r = 0; r < 8; ++r){
                int m = rowbase + ms*16 + 8*g + r;
                if(m < M_valid) O[(size_t)m*N + n] = acc[ms][r] + bb;
            }
    } else if constexpr (EPI == 1){
        float* O = (float*)outp; int h = n >> 6, d = n & 63;
        #pragma unroll
        for(int ms = 0; ms < 4; ++ms)
            #pragma unroll
            for(int r = 0; r < 8; ++r){
                int m = rowbase + ms*16 + 8*g + r;
                if(m < M_valid){
                    int b2 = m / seqlen, s = m % seqlen;
                    O[(((size_t)(b2*HEADS + h))*seqlen + s)*HEAD_DIM + d] = acc[ms][r];
                }
            }
    } else {
        f16_t* O = (f16_t*)outp; float bb = bias[n];
        #pragma unroll
        for(int ms = 0; ms < 4; ++ms)
            #pragma unroll
            for(int r = 0; r < 8; ++r){
                int m = rowbase + ms*16 + 8*g + r;
                float x = acc[ms][r] + bb;
                float t = tanhf(0.7978845608028654f*(x + 0.044715f*x*x*x));
                O[(size_t)m*N + n] = (f16_t)(0.5f*x*(1.f + t));
            }
    }
}

// ------------------- QK head-layernorm + RoPE + emit padded f16 Q/K and V^T
__global__ void k_qkv_post(float* __restrict__ Q, float* __restrict__ K, const float* __restrict__ V,
                           const float* __restrict__ nqw, const float* __restrict__ nqb,
                           const float* __restrict__ nkw, const float* __restrict__ nkb,
                           const float* __restrict__ rc, const float* __restrict__ rs,
                           f16_t* Qh, f16_t* Kh, f16_t* Vt,
                           int seqlen, int rope_from, int spad){
    __shared__ float r1[64], r2[64], rq[64], rk[64];
    int s = blockIdx.x, h = blockIdx.y, b = blockIdx.z;
    int d = threadIdx.x;
    size_t base = (((size_t)(b*HEADS + h))*seqlen + s)*64;
    float q = Q[base + d], k = K[base + d];
    r1[d] = q; r2[d] = q*q; __syncthreads();
    for(int st = 32; st > 0; st >>= 1){ if(d < st){ r1[d] += r1[d+st]; r2[d] += r2[d+st]; } __syncthreads(); }
    float muq = r1[0]*(1.f/64.f), varq = r2[0]*(1.f/64.f) - muq*muq; __syncthreads();
    r1[d] = k; r2[d] = k*k; __syncthreads();
    for(int st = 32; st > 0; st >>= 1){ if(d < st){ r1[d] += r1[d+st]; r2[d] += r2[d+st]; } __syncthreads(); }
    float muk = r1[0]*(1.f/64.f), vark = r2[0]*(1.f/64.f) - muk*muk; __syncthreads();
    float qn = (q - muq)*rsqrtf(varq + 1e-6f)*nqw[d] + nqb[d];
    float kn = (k - muk)*rsqrtf(vark + 1e-6f)*nkw[d] + nkb[d];
    rq[d] = qn; rk[d] = kn; __syncthreads();
    if(s >= rope_from){
        int p = s - rope_from;
        float rotq = (d & 1) ? rq[d-1] : -rq[d+1];
        float rotk = (d & 1) ? rk[d-1] : -rk[d+1];
        float c = rc[(size_t)p*64 + d], sn = rs[(size_t)p*64 + d];
        qn = qn*c + rotq*sn;
        kn = kn*c + rotk*sn;
    }
    Q[base + d] = qn; K[base + d] = kn;
    if(Qh){
        size_t pb = (((size_t)(b*HEADS + h))*spad + s)*64 + d;
        Qh[pb] = (f16_t)qn; Kh[pb] = (f16_t)kn;
        Vt[(((size_t)(b*HEADS + h))*64 + d)*spad + s] = (f16_t)V[base + d];
    }
}

// --------------------------------------------------- flash SDPA (WMMA based)
__global__ __launch_bounds__(128)
void k_flash(const f16_t* __restrict__ Qh, const f16_t* __restrict__ Kh,
             const f16_t* __restrict__ Vt, float* __restrict__ Of){
    __shared__ alignas(32) f16_t lsP[4][16*32];
    const int wv = threadIdx.x >> 5, lane = threadIdx.x & 31;
    const int g = lane >> 4, l15 = lane & 15;
    const int qtile = blockIdx.x*4 + wv;
    if(qtile >= QTILES) return;
    const size_t bh = (size_t)blockIdx.z*HEADS + blockIdx.y;
    const f16_t* Qp = Qh + bh*(size_t)S_PAD*HEAD_DIM;
    const f16_t* Kp = Kh + bh*(size_t)S_PAD*HEAD_DIM;
    const f16_t* Vp = Vt + bh*(size_t)HEAD_DIM*S_PAD;
    const int qbase = qtile*16;
    v16h aQ0, aQ1;
    {
        const f16_t* qr = Qp + (size_t)(qbase + l15)*HEAD_DIM + 16*g;
        v8h lo = *(const v8h*)qr;        v8h hi = *(const v8h*)(qr + 8);
        aQ0 = SHUF16(lo, hi);
        lo = *(const v8h*)(qr + 32);     hi = *(const v8h*)(qr + 40);
        aQ1 = SHUF16(lo, hi);
    }
    v8f o0, o1, o2, o3;
    #pragma unroll
    for(int i = 0; i < 8; ++i){ o0[i]=0.f; o1[i]=0.f; o2[i]=0.f; o3[i]=0.f; }
    float rm[8], rl[8];
    #pragma unroll
    for(int r = 0; r < 8; ++r){ rm[r] = -3e38f; rl[r] = 0.f; }
    f16_t* P = &lsP[wv][0];
    for(int ks = 0; ks < S_PAD; ks += 32){
        v8f s0v, s1v;
        #pragma unroll
        for(int i = 0; i < 8; ++i){ s0v[i]=0.f; s1v[i]=0.f; }
        {
            v16h b = *(const v16h*)(Kp + (size_t)(ks + l15)*HEAD_DIM + 16*g);
            s0v = wmma16(aQ0, b, s0v);
            b = *(const v16h*)(Kp + (size_t)(ks + l15)*HEAD_DIM + 16*g + 32);
            s0v = wmma16(aQ1, b, s0v);
            b = *(const v16h*)(Kp + (size_t)(ks + 16 + l15)*HEAD_DIM + 16*g);
            s1v = wmma16(aQ0, b, s1v);
            b = *(const v16h*)(Kp + (size_t)(ks + 16 + l15)*HEAD_DIM + 16*g + 32);
            s1v = wmma16(aQ1, b, s1v);
        }
        const int key0 = ks + l15, key1 = ks + 16 + l15;
        float p0[8], p1[8], mnew[8];
        #pragma unroll
        for(int r = 0; r < 8; ++r){
            float a0 = (key0 < S_TOT) ? s0v[r]*0.125f : -1e30f;
            float a1 = (key1 < S_TOT) ? s1v[r]*0.125f : -1e30f;
            p0[r] = a0; p1[r] = a1;
            float mx = fmaxf(a0, a1);
            mx = fmaxf(mx, __shfl_xor(mx, 1, 32));
            mx = fmaxf(mx, __shfl_xor(mx, 2, 32));
            mx = fmaxf(mx, __shfl_xor(mx, 4, 32));
            mx = fmaxf(mx, __shfl_xor(mx, 8, 32));
            mnew[r] = fmaxf(rm[r], mx);
        }
        #pragma unroll
        for(int r = 0; r < 8; ++r){
            float scl = __expf(rm[r] - mnew[r]);
            rm[r] = mnew[r];
            float e0 = __expf(p0[r] - mnew[r]);
            float e1 = __expf(p1[r] - mnew[r]);
            float sum = e0 + e1;
            sum += __shfl_xor(sum, 1, 32);
            sum += __shfl_xor(sum, 2, 32);
            sum += __shfl_xor(sum, 4, 32);
            sum += __shfl_xor(sum, 8, 32);
            rl[r] = rl[r]*scl + sum;
            o0[r] *= scl; o1[r] *= scl; o2[r] *= scl; o3[r] *= scl;
            P[(r + 8*g)*32 + l15]      = (f16_t)e0;
            P[(r + 8*g)*32 + 16 + l15] = (f16_t)e1;
        }
        v8h plo = *(const v8h*)(P + l15*32 + 16*g);
        v8h phi = *(const v8h*)(P + l15*32 + 16*g + 8);
        v16h aP = SHUF16(plo, phi);
        v16h bv;
        bv = *(const v16h*)(Vp + (size_t)( 0 + l15)*S_PAD + ks + 16*g); o0 = wmma16(aP, bv, o0);
        bv = *(const v16h*)(Vp + (size_t)(16 + l15)*S_PAD + ks + 16*g); o1 = wmma16(aP, bv, o1);
        bv = *(const v16h*)(Vp + (size_t)(32 + l15)*S_PAD + ks + 16*g); o2 = wmma16(aP, bv, o2);
        bv = *(const v16h*)(Vp + (size_t)(48 + l15)*S_PAD + ks + 16*g); o3 = wmma16(aP, bv, o3);
    }
    #pragma unroll
    for(int r = 0; r < 8; ++r){
        int s = qbase + 8*g + r;
        if(s < S_TOT){
            float inv = 1.f/rl[r];
            size_t ob = (bh*(size_t)S_TOT + s)*64;
            Of[ob +      l15] = o0[r]*inv;
            Of[ob + 16 + l15] = o1[r]*inv;
            Of[ob + 32 + l15] = o2[r]*inv;
            Of[ob + 48 + l15] = o3[r]*inv;
        }
    }
}

// -------------------------- merge heads (b,h,s,d)f32 -> row-major f16, zero pad
__global__ void k_merge(const float* __restrict__ X, f16_t* __restrict__ Y,
                        int seqlen, int Mvalid, int Mpad){
    size_t i = (size_t)blockIdx.x*256 + threadIdx.x;
    if(i >= (size_t)Mpad*DIM) return;
    int m = (int)(i / DIM), n = (int)(i % DIM);
    float v = 0.f;
    if(m < Mvalid){
        int b = m / seqlen, s = m % seqlen, h = n >> 6, d = n & 63;
        v = X[(((size_t)(b*HEADS + h))*seqlen + s)*64 + d];
    }
    Y[i] = (f16_t)v;
}

// ----------------------------------------------------------- concept attention
__global__ void k_cscore(const float* __restrict__ Cq, const float* __restrict__ Ck,
                         const float* __restrict__ Kf, float* __restrict__ cs){
    __shared__ float q[8*64];
    size_t bh = (size_t)blockIdx.z*HEADS + blockIdx.y;
    for(int i = threadIdx.x; i < 512; i += 256) q[i] = Cq[bh*512 + i];
    __syncthreads();
    int j = blockIdx.x*256 + threadIdx.x;
    if(j >= S_CAT) return;
    const float* kv = (j < NCPT) ? (Ck + (bh*NCPT + j)*64)
                                 : (Kf + (bh*S_TOT + TEXT_LEN + (j - NCPT))*64);
    float acc[8];
    #pragma unroll
    for(int t = 0; t < 8; ++t) acc[t] = 0.f;
    for(int d = 0; d < 64; ++d){
        float kk = kv[d];
        #pragma unroll
        for(int t = 0; t < 8; ++t) acc[t] += q[t*64 + d]*kk;
    }
    #pragma unroll
    for(int t = 0; t < 8; ++t) cs[(bh*NCPT + t)*S_CAT + j] = acc[t]*0.125f;
}

__global__ void k_csoftmax(float* __restrict__ cs){
    __shared__ float red[256];
    size_t bh = (size_t)blockIdx.z*HEADS + blockIdx.y;
    float* row = cs + (bh*NCPT + blockIdx.x)*S_CAT;
    float mx = -3e38f;
    for(int j = threadIdx.x; j < S_CAT; j += 256) mx = fmaxf(mx, row[j]);
    red[threadIdx.x] = mx; __syncthreads();
    for(int st = 128; st > 0; st >>= 1){ if(threadIdx.x < st) red[threadIdx.x] = fmaxf(red[threadIdx.x], red[threadIdx.x+st]); __syncthreads(); }
    mx = red[0]; __syncthreads();
    float sum = 0.f;
    for(int j = threadIdx.x; j < S_CAT; j += 256){ float e = __expf(row[j] - mx); row[j] = e; sum += e; }
    red[threadIdx.x] = sum; __syncthreads();
    for(int st = 128; st > 0; st >>= 1){ if(threadIdx.x < st) red[threadIdx.x] += red[threadIdx.x+st]; __syncthreads(); }
    float inv = 1.f/red[0];
    for(int j = threadIdx.x; j < S_CAT; j += 256) row[j] *= inv;
}

__global__ void k_cpv(const float* __restrict__ cs, const float* __restrict__ Cv,
                      const float* __restrict__ Vf, float* __restrict__ Cattn){
    size_t bh = (size_t)blockIdx.y*HEADS + blockIdx.x;
    int q = threadIdx.x >> 6, d = threadIdx.x & 63;
    const float* p = cs + (bh*NCPT + q)*S_CAT;
    float acc = 0.f;
    for(int j = 0; j < S_CAT; ++j){
        float v = (j < NCPT) ? Cv[(bh*NCPT + j)*64 + d]
                             : Vf[(bh*S_TOT + TEXT_LEN + j - NCPT)*64 + d];
        acc += p[j]*v;
    }
    Cattn[(bh*NCPT + q)*64 + d] = acc;
}

// ----------------------------------------------------------- analysis kernels
__global__ void k_hscore(const float* __restrict__ Of, float* __restrict__ hs){
    __shared__ float r1[256], r2[256];
    int h = blockIdx.x;
    float s = 0.f, s2 = 0.f;
    for(int p = threadIdx.x; p < IMG_LEN; p += 256){
        const float* row = Of + (((size_t)(HEADS + h))*S_TOT + TEXT_LEN + p)*64;
        float m = 0.f;
        for(int d = 0; d < 64; ++d) m += row[d];
        m *= (1.f/64.f);
        s += m; s2 += m*m;
    }
    r1[threadIdx.x] = s; r2[threadIdx.x] = s2; __syncthreads();
    for(int st = 128; st > 0; st >>= 1){ if(threadIdx.x < st){ r1[threadIdx.x] += r1[threadIdx.x+st]; r2[threadIdx.x] += r2[threadIdx.x+st]; } __syncthreads(); }
    if(threadIdx.x == 0){
        float mu = r1[0]*(1.f/IMG_LEN);
        float var = r2[0]*(1.f/IMG_LEN) - mu*mu;
        hs[h] = sqrtf(fmaxf(var, 0.f));
    }
}

__global__ void k_top4(const float* __restrict__ hs, int* __restrict__ sel){
    if(threadIdx.x == 0 && blockIdx.x == 0){
        bool used[16]; for(int h = 0; h < 16; ++h) used[h] = false;
        for(int k = 0; k < 4; ++k){
            float best = -3e38f; int bi = 0;
            for(int h = 0; h < 16; ++h) if(!used[h] && hs[h] > best){ best = hs[h]; bi = h; }
            used[bi] = true; sel[k] = bi;
        }
    }
}

__global__ void k_crossmaps(const float* __restrict__ Qf, const float* __restrict__ Ck,
                            float* __restrict__ out){
    int p = blockIdx.x*128 + threadIdx.x;
    int c = blockIdx.y;
    float acc = 0.f;
    for(int h = 0; h < 16; ++h){
        const float* qp = Qf + (((size_t)(HEADS + h))*S_TOT + TEXT_LEN + p)*64;
        const float* kp = Ck + ((size_t)(HEADS + h)*NCPT + c)*64;
        for(int d = 0; d < 64; ++d) acc += qp[d]*kp[d];
    }
    out[(size_t)c*IMG_LEN + p] = acc*(1.f/16.f);
}

__global__ void k_conceptmaps(const float* __restrict__ Cattn, const float* __restrict__ Of,
                              float* __restrict__ out){
    int p = blockIdx.x*128 + threadIdx.x;
    int c = blockIdx.y;
    float acc = 0.f;
    for(int h = 0; h < 16; ++h){
        const float* ap = Cattn + ((size_t)(HEADS + h)*NCPT + c)*64;
        const float* ip = Of + (((size_t)(HEADS + h))*S_TOT + TEXT_LEN + p)*64;
        for(int d = 0; d < 64; ++d) acc += ap[d]*ip[d];
    }
    out[(size_t)c*IMG_LEN + p] = acc;
}

__global__ void k_simsel(const float* __restrict__ Qf, const float* __restrict__ Ck,
                         const float* __restrict__ Of, float* __restrict__ imapw){
    __shared__ float ck[64], sel[64];
    __shared__ float bv[128]; __shared__ int bi[128];
    int f = blockIdx.x, t = blockIdx.y, h = blockIdx.z;
    size_t bh = (size_t)HEADS + h;
    for(int i = threadIdx.x; i < 64; i += 128) ck[i] = Ck[(bh*NCPT + t)*64 + i];
    __syncthreads();
    float best = -3e38f; int besti = 0;
    for(int p = threadIdx.x; p < PATCHES; p += 128){
        const float* qp = Qf + (bh*S_TOT + TEXT_LEN + f*PATCHES + p)*64;
        float acc = 0.f;
        for(int d = 0; d < 64; ++d) acc += qp[d]*ck[d];
        if(acc > best){ best = acc; besti = p; }
    }
    bv[threadIdx.x] = best; bi[threadIdx.x] = besti; __syncthreads();
    for(int st = 64; st > 0; st >>= 1){
        if(threadIdx.x < st){
            float ov = bv[threadIdx.x+st]; int oi = bi[threadIdx.x+st];
            if(ov > bv[threadIdx.x] || (ov == bv[threadIdx.x] && oi < bi[threadIdx.x])){
                bv[threadIdx.x] = ov; bi[threadIdx.x] = oi;
            }
        }
        __syncthreads();
    }
    int idx = bi[0];
    for(int i = threadIdx.x; i < 64; i += 128)
        sel[i] = Of[(bh*S_TOT + TEXT_LEN + f*PATCHES + idx)*64 + i];
    __syncthreads();
    for(int p = threadIdx.x; p < PATCHES; p += 128){
        const float* op = Of + (bh*S_TOT + TEXT_LEN + f*PATCHES + p)*64;
        float acc = 0.f;
        for(int d = 0; d < 64; ++d) acc += sel[d]*op[d];
        imapw[(((size_t)h*NCPT + t)*FRAMES + f)*PATCHES + p] = acc;
    }
}

__global__ void k_imapfin(const float* __restrict__ imapw, const int* __restrict__ sel,
                          float* __restrict__ out_sel, float* __restrict__ out_sum){
    int fp = blockIdx.x*256 + threadIdx.x;
    if(fp >= IMG_LEN) return;
    int s0 = sel[0], s1 = sel[1], s2 = sel[2], s3 = sel[3];
    float accS[8], accA[8];
    #pragma unroll
    for(int t = 0; t < 8; ++t){ accS[t] = 0.f; accA[t] = 0.f; }
    for(int h = 0; h < 16; ++h){
        float v[8]; float mu = 0.f;
        #pragma unroll
        for(int t = 0; t < 8; ++t){ v[t] = imapw[((size_t)h*NCPT + t)*IMG_LEN + fp]; mu += v[t]; }
        mu *= 0.125f;
        float var = 0.f;
        #pragma unroll
        for(int t = 0; t < 8; ++t){ float dd = v[t]-mu; var += dd*dd; }
        float sd = sqrtf(var*(1.f/7.f)) + 1e-6f;
        bool issel = (h == s0) | (h == s1) | (h == s2) | (h == s3);
        #pragma unroll
        for(int t = 0; t < 8; ++t){
            float nv = (v[t]-mu)/sd;
            accA[t] += nv;
            if(issel) accS[t] += nv;
        }
    }
    #pragma unroll
    for(int t = 0; t < 8; ++t){
        out_sel[(size_t)t*IMG_LEN + fp] = accS[t]*0.25f;
        out_sum[(size_t)t*IMG_LEN + fp] = accA[t];
    }
}

// ------------------------------------------------------------------ residuals
__global__ void k_resid(float* __restrict__ dst, const float* __restrict__ x0,
                        const float* __restrict__ sbuf, int gate_c,
                        const float* __restrict__ src, int S, int src_brow, int src_roff){
    size_t i = (size_t)blockIdx.x*256 + threadIdx.x;
    size_t tot = (size_t)NB*S*DIM;
    if(i >= tot) return;
    int n = (int)(i % DIM);
    int r = (int)((i / DIM) % S);
    int b = (int)(i / ((size_t)S*DIM));
    dst[i] = x0[i] + sbuf[(size_t)b*6*DIM + gate_c*DIM + n]
                   * src[((size_t)(b*src_brow + src_roff + r))*DIM + n];
}

// ============================================================== host launcher
static inline size_t aup(size_t x){ return (x + 255) & ~(size_t)255; }
static inline int cdiv(size_t a, size_t b){ return (int)((a + b - 1)/b); }

extern "C" void kernel_launch(void* const* d_in, const int* in_sizes, int n_in,
                              void* d_out, int out_size, void* d_ws, size_t ws_size,
                              hipStream_t stream){
    (void)in_sizes; (void)n_in; (void)out_size; (void)ws_size;
    const float* hid  = (const float*)d_in[0];
    const float* enc  = (const float*)d_in[1];
    const float* conc = (const float*)d_in[2];
    const float* temb = (const float*)d_in[3];
    const float* rc   = (const float*)d_in[4];
    const float* rs   = (const float*)d_in[5];
    const float* n1w  = (const float*)d_in[6];
    const float* n1b  = (const float*)d_in[7];
    const float* l1w  = (const float*)d_in[8];
    const float* l1b  = (const float*)d_in[9];
    const float* n2w  = (const float*)d_in[10];
    const float* n2b  = (const float*)d_in[11];
    const float* l2w  = (const float*)d_in[12];
    const float* l2b  = (const float*)d_in[13];
    const float* wq   = (const float*)d_in[14];
    const float* wk   = (const float*)d_in[15];
    const float* wvp  = (const float*)d_in[16];
    const float* nqw  = (const float*)d_in[17];
    const float* nqb  = (const float*)d_in[18];
    const float* nkw  = (const float*)d_in[19];
    const float* nkb  = (const float*)d_in[20];
    const float* wo   = (const float*)d_in[21];
    const float* bo   = (const float*)d_in[22];
    const float* fw1  = (const float*)d_in[23];
    const float* fb1  = (const float*)d_in[24];
    const float* fw2  = (const float*)d_in[25];
    const float* fb2  = (const float*)d_in[26];

    float* out = (float*)d_out;
    const size_t OFF_H   = 0;
    const size_t OFF_E   = (size_t)NB*IMG_LEN*DIM;
    const size_t OFF_C   = OFF_E + (size_t)NB*TEXT_LEN*DIM;
    const size_t OFF_SEL = OFF_C + (size_t)NB*NCPT*DIM;
    const size_t OFF_IM  = OFF_SEL + (size_t)NCPT*IMG_LEN;
    const size_t OFF_CM  = OFF_IM  + (size_t)NCPT*IMG_LEN;
    const size_t OFF_CR  = OFF_CM  + (size_t)NCPT*IMG_LEN;

    char* w = (char*)d_ws; size_t off = 0;
    auto AF = [&](size_t n)->float*{ float* p = (float*)(w + off); off = aup(off + n*4); return p; };
    auto AH = [&](size_t n)->f16_t*{ f16_t* p = (f16_t*)(w + off); off = aup(off + n*2); return p; };
    auto AI = [&](size_t n)->int*  { int*   p = (int*)(w + off);   off = aup(off + n*4); return p; };

    float* s1    = AF((size_t)NB*6*DIM);
    float* s2    = AF((size_t)NB*6*DIM);
    float* Qf    = AF((size_t)NB*HEADS*S_TOT*64);
    float* Kf    = AF((size_t)NB*HEADS*S_TOT*64);
    float* Vf    = AF((size_t)NB*HEADS*S_TOT*64);
    float* Of    = AF((size_t)NB*HEADS*S_TOT*64);
    float* Cqf   = AF((size_t)NB*HEADS*NCPT*64);
    float* Ckf   = AF((size_t)NB*HEADS*NCPT*64);
    float* Cvf   = AF((size_t)NB*HEADS*NCPT*64);
    float* Catt  = AF((size_t)NB*HEADS*NCPT*64);
    float* proj  = AF((size_t)M_PAD*DIM);
    float* cproj = AF((size_t)16*DIM);
    float* ff2f  = AF((size_t)M_PAD*DIM);
    float* ffc2f = AF((size_t)16*DIM);
    float* csc   = AF((size_t)NB*HEADS*NCPT*S_CAT);
    float* hsc   = AF(16);
    float* imapw = AF((size_t)HEADS*NCPT*IMG_LEN);
    int*   selidx= AI(4);
    f16_t* xs16  = AH((size_t)M_PAD*DIM);
    f16_t* nc16  = AH((size_t)CM_PAD*DIM);
    f16_t* xs2   = AH((size_t)M_PAD*DIM);
    f16_t* nc2   = AH((size_t)CM_PAD*DIM);
    f16_t* wq16  = AH((size_t)DIM*DIM);
    f16_t* wk16  = AH((size_t)DIM*DIM);
    f16_t* wv16  = AH((size_t)DIM*DIM);
    f16_t* wo16  = AH((size_t)DIM*DIM);
    f16_t* fw116 = AH((size_t)DIM*FF_DIM);
    f16_t* fw216 = AH((size_t)FF_DIM*DIM);
    f16_t* Qh    = AH((size_t)NB*HEADS*S_PAD*64);
    f16_t* Kh    = AH((size_t)NB*HEADS*S_PAD*64);
    f16_t* Vt    = AH((size_t)NB*HEADS*64*S_PAD);
    f16_t* Oh16  = AH((size_t)M_PAD*DIM);
    f16_t* Ch16  = AH((size_t)CM_PAD*DIM);
    f16_t* ff116 = AH((size_t)M_PAD*FF_DIM);
    f16_t* ffc116= AH((size_t)CM_PAD*FF_DIM);

    // ---- weights to f16
    size_t nsq = (size_t)DIM*DIM, nff = (size_t)DIM*FF_DIM;
    k_f32_to_f16<<<cdiv(nsq,256),256,0,stream>>>(wq,  wq16,  nsq);
    k_f32_to_f16<<<cdiv(nsq,256),256,0,stream>>>(wk,  wk16,  nsq);
    k_f32_to_f16<<<cdiv(nsq,256),256,0,stream>>>(wvp, wv16,  nsq);
    k_f32_to_f16<<<cdiv(nsq,256),256,0,stream>>>(wo,  wo16,  nsq);
    k_f32_to_f16<<<cdiv(nff,256),256,0,stream>>>(fw1, fw116, nff);
    k_f32_to_f16<<<cdiv(nff,256),256,0,stream>>>(fw2, fw216, nff);

    // ---- adaLN vectors
    k_modproj<<<dim3(24,NB),256,0,stream>>>(temb, n1w, n1b, s1);
    k_modproj<<<dim3(24,NB),256,0,stream>>>(temb, n2w, n2b, s2);

    // ---- zero fills (pads + padded attn buffers)
    size_t npad = (size_t)(M_PAD - M_ROWS)*DIM;
    k_fill_h<<<cdiv(npad,256),256,0,stream>>>(xs16 + (size_t)M_ROWS*DIM, npad);
    k_fill_h<<<cdiv(npad,256),256,0,stream>>>(xs2  + (size_t)M_ROWS*DIM, npad);
    size_t ncp = (size_t)CM_PAD*DIM;
    k_fill_h<<<cdiv(ncp,256),256,0,stream>>>(nc16, ncp);
    k_fill_h<<<cdiv(ncp,256),256,0,stream>>>(nc2,  ncp);
    size_t nqk = (size_t)NB*HEADS*S_PAD*64;
    k_fill_h<<<cdiv(nqk,256),256,0,stream>>>(Qh, nqk);
    k_fill_h<<<cdiv(nqk,256),256,0,stream>>>(Kh, nqk);
    k_fill_h<<<cdiv(nqk,256),256,0,stream>>>(Vt, nqk);

    // ---- LN + modulation (pass 1)
    k_ln_mod<<<dim3(TEXT_LEN,NB),256,0,stream>>>(enc,  s1, l1w, l1b, xs16, TEXT_LEN, 3,4, 0,        S_TOT);
    k_ln_mod<<<dim3(IMG_LEN ,NB),256,0,stream>>>(hid,  s1, l1w, l1b, xs16, IMG_LEN,  0,1, TEXT_LEN, S_TOT);
    k_ln_mod<<<dim3(NCPT    ,NB),256,0,stream>>>(conc, s1, l1w, l1b, nc16, NCPT,     3,4, 0,        NCPT);

    // ---- QKV projections (WMMA, 64x64 tiles)
    dim3 gQKV(DIM/64, M_PAD/64);
    k_gemm<1><<<gQKV,128,0,stream>>>(xs16, wq16, nullptr, Qf, M_ROWS, DIM, DIM, S_TOT);
    k_gemm<1><<<gQKV,128,0,stream>>>(xs16, wk16, nullptr, Kf, M_ROWS, DIM, DIM, S_TOT);
    k_gemm<1><<<gQKV,128,0,stream>>>(xs16, wv16, nullptr, Vf, M_ROWS, DIM, DIM, S_TOT);
    dim3 gC(DIM/64, 1);
    k_gemm<1><<<gC,128,0,stream>>>(nc16, wq16, nullptr, Cqf, 16, DIM, DIM, NCPT);
    k_gemm<1><<<gC,128,0,stream>>>(nc16, wk16, nullptr, Ckf, 16, DIM, DIM, NCPT);
    k_gemm<1><<<gC,128,0,stream>>>(nc16, wv16, nullptr, Cvf, 16, DIM, DIM, NCPT);

    // ---- QK norm + RoPE + padded f16 / V^T
    k_qkv_post<<<dim3(S_TOT,HEADS,NB),64,0,stream>>>(Qf, Kf, Vf, nqw, nqb, nkw, nkb, rc, rs,
                                                     Qh, Kh, Vt, S_TOT, TEXT_LEN, S_PAD);
    k_qkv_post<<<dim3(NCPT,HEADS,NB),64,0,stream>>>(Cqf, Ckf, Cvf, nqw, nqb, nkw, nkb, rc, rs,
                                                    nullptr, nullptr, nullptr, NCPT, 1<<30, S_PAD);

    // ---- main SDPA (flash, WMMA)
    k_flash<<<dim3((QTILES+3)/4, HEADS, NB),128,0,stream>>>(Qh, Kh, Vt, Of);
    k_merge<<<cdiv((size_t)M_PAD*DIM,256),256,0,stream>>>(Of, Oh16, S_TOT, M_ROWS, M_PAD);

    // ---- concept attention (8 queries)
    k_cscore  <<<dim3(cdiv(S_CAT,256), HEADS, NB),256,0,stream>>>(Cqf, Ckf, Kf, csc);
    k_csoftmax<<<dim3(NCPT, HEADS, NB),256,0,stream>>>(csc);
    k_cpv     <<<dim3(HEADS, NB),512,0,stream>>>(csc, Cvf, Vf, Catt);
    k_merge<<<cdiv((size_t)CM_PAD*DIM,256),256,0,stream>>>(Catt, Ch16, NCPT, 16, CM_PAD);

    // ---- output projections
    k_gemm<0><<<dim3(DIM/64, M_PAD/64),128,0,stream>>>(Oh16, wo16, bo, proj,  M_ROWS, DIM, DIM, 0);
    k_gemm<0><<<dim3(DIM/64, 1),128,0,stream>>>(Ch16, wo16, bo, cproj, 16, DIM, DIM, 0);

    // ---- analysis outputs (batch 1)
    k_hscore<<<16,256,0,stream>>>(Of, hsc);
    k_top4<<<1,32,0,stream>>>(hsc, selidx);
    k_crossmaps  <<<dim3(IMG_LEN/128, NCPT),128,0,stream>>>(Qf, Ckf, out + OFF_CR);
    k_conceptmaps<<<dim3(IMG_LEN/128, NCPT),128,0,stream>>>(Catt, Of, out + OFF_CM);
    k_simsel<<<dim3(FRAMES, NCPT, HEADS),128,0,stream>>>(Qf, Ckf, Of, imapw);
    k_imapfin<<<cdiv(IMG_LEN,256),256,0,stream>>>(imapw, selidx, out + OFF_SEL, out + OFF_IM);

    // ---- residual 1 (into d_out)
    k_resid<<<cdiv((size_t)NB*IMG_LEN*DIM,256),256,0,stream>>>(out+OFF_H, hid,  s1, 2, proj,  IMG_LEN,  S_TOT, TEXT_LEN);
    k_resid<<<cdiv((size_t)NB*TEXT_LEN*DIM,256),256,0,stream>>>(out+OFF_E, enc,  s1, 5, proj,  TEXT_LEN, S_TOT, 0);
    k_resid<<<cdiv((size_t)NB*NCPT*DIM,256),256,0,stream>>>(out+OFF_C, conc, s1, 5, cproj, NCPT,     NCPT,  0);

    // ---- LN + modulation (pass 2)
    k_ln_mod<<<dim3(TEXT_LEN,NB),256,0,stream>>>(out+OFF_E, s2, l2w, l2b, xs2, TEXT_LEN, 3,4, 0,        S_TOT);
    k_ln_mod<<<dim3(IMG_LEN ,NB),256,0,stream>>>(out+OFF_H, s2, l2w, l2b, xs2, IMG_LEN,  0,1, TEXT_LEN, S_TOT);
    k_ln_mod<<<dim3(NCPT    ,NB),256,0,stream>>>(out+OFF_C, s2, l2w, l2b, nc2, NCPT,     3,4, 0,        NCPT);

    // ---- FFN (WMMA)
    k_gemm<2><<<dim3(FF_DIM/64, M_PAD/64),128,0,stream>>>(xs2, fw116, fb1, ff116,  M_PAD, FF_DIM, DIM, 0);
    k_gemm<2><<<dim3(FF_DIM/64, 1),128,0,stream>>>(nc2, fw116, fb1, ffc116, CM_PAD, FF_DIM, DIM, 0);
    k_gemm<0><<<dim3(DIM/64, M_PAD/64),128,0,stream>>>(ff116,  fw216, fb2, ff2f,  M_ROWS, DIM, FF_DIM, 0);
    k_gemm<0><<<dim3(DIM/64, 1),128,0,stream>>>(ffc116, fw216, fb2, ffc2f, 16,    DIM, FF_DIM, 0);

    // ---- residual 2 (in place on d_out)
    k_resid<<<cdiv((size_t)NB*IMG_LEN*DIM,256),256,0,stream>>>(out+OFF_H, out+OFF_H, s2, 2, ff2f,  IMG_LEN,  S_TOT, TEXT_LEN);
    k_resid<<<cdiv((size_t)NB*TEXT_LEN*DIM,256),256,0,stream>>>(out+OFF_E, out+OFF_E, s2, 5, ff2f,  TEXT_LEN, S_TOT, 0);
    k_resid<<<cdiv((size_t)NB*NCPT*DIM,256),256,0,stream>>>(out+OFF_C, out+OFF_C, s2, 5, ffc2f, NCPT,     NCPT,  0);
}
